// Zentorch_Simulated_Woq_Linear_51488067945062
// MI455X (gfx1250) — matrix operand order repaired and checked by reading, not compiled, not run on hardware
//
#include <hip/hip_runtime.h>
#include <hip/hip_bf16.h>
#include <stdint.h>

// AWQ int4 weight-only quant GEMM for gfx1250 (MI455X).
// out[M, N] = X[M, K] * dequant(qweight)[K, N] + bias, bf16 in/out.
// M = B*T = 8192, K = 4096, N = 11008, group G = 128 (compile-time).
//
// 256x128 block tile, BK=32 K-steps, v_wmma_f32_16x16x32_bf16,
// 8 waves in 4(M) x 2(N) grid, 64x64 per wave (16 accumulators).
// Double-buffered LDS ping-pong, one barrier per K-step:
//   - A tile staged with GLOBAL_LOAD_ASYNC_TO_LDS_B128 (ASYNCcnt path)
//   - B tile: qweight/scales pulled to registers one step ahead; int4
//     dequant uses shift-based sign extension and packs two k-values
//     per ds_store_b32, overlapped behind the WMMAs.

typedef __attribute__((ext_vector_type(16))) __bf16 v16bf;
typedef __attribute__((ext_vector_type(8)))  float  v8f;

#define BM 256
#define BN 128
#define BK 32
#define KDIM 4096
#define NDIM 11008
#define NPACKS (NDIM / 8)          // 1376 int32 columns
#define KTILES (KDIM / BK)         // 128

__device__ __forceinline__ unsigned short f32_to_bf16(float f) {
    unsigned u = __float_as_uint(f);
    unsigned r = u + 0x7FFFu + ((u >> 16) & 1u);   // round-to-nearest-even
    return (unsigned short)(r >> 16);
}
__device__ __forceinline__ float bf16_to_f32(unsigned short h) {
    return __uint_as_float(((unsigned)h) << 16);
}

__global__ __launch_bounds__(256, 1)
void woq_awq_gemm_bf16(const unsigned short* __restrict__ X,    // [M, K] bf16 bits
                       const int*            __restrict__ QW,   // [K, N/8]
                       const float*          __restrict__ S,    // [K/128, N]
                       const unsigned short* __restrict__ bias, // [N] bf16 bits
                       unsigned short*       __restrict__ out)  // [M, N] bf16 bits
{
    __shared__ unsigned short As[2][BM][BK];    // activations   [buf][m][k]  32 KB
    __shared__ unsigned short Bs[2][BN][BK];    // dequant wgts  [buf][n][k]  16 KB

    const int tid  = threadIdx.x;
    const int lane = tid & 31;
    const int wave = tid >> 5;
    const int wm   = (wave >> 1) * 64;          // wave M offset: 0,64,128,192
    const int wn   = (wave & 1)  * 64;          // wave N offset: 0,64
    const int row0 = blockIdx.y * BM;
    const int col0 = blockIdx.x * BN;

    // A staging: 1024 x 16B chunks per tile -> 4 async DMAs per thread.
    const int ar = tid >> 2;                    // A row base (passes add 64)
    const int ac = (tid & 3) * 8;               // A col (8 bf16 = 16B)
    // B staging: thread owns k-pair (kk2, kk2+1) for 8 consecutive n.
    const int bp  = tid & 15;                   // int32 column within tile
    const int kk2 = (tid >> 4) * 2;             // even k within tile, 0..30

    const uint64_t Xbase  = (uint64_t)X;
    const unsigned asBase = (unsigned)(uintptr_t)&As[0][0][0];  // LDS byte offset

    unsigned aoffg[4], aoffl[4];
    #pragma unroll
    for (int p = 0; p < 4; ++p) {
        aoffg[p] = (unsigned)(((row0 + ar + p * 64) * KDIM + ac) * 2);
        aoffl[p] = (unsigned)(((ar + p * 64) * BK + ac) * 2);
    }

    // Async DMA of one A tile (4 x b128 per lane) into LDS buffer `buf`.
    auto stageA = [&](int k0, int buf) {
        unsigned lbase = asBase + (unsigned)(buf * BM * BK * 2);
        #pragma unroll
        for (int p = 0; p < 4; ++p) {
            unsigned l = lbase + aoffl[p];
            unsigned g = aoffg[p] + (unsigned)(k0 * 2);
            asm volatile("global_load_async_to_lds_b128 %0, %1, %2"
                         :: "v"(l), "v"(g), "s"(Xbase) : "memory");
        }
    };

    // Pull next tile's qweight + scales into registers (in flight during WMMAs).
    auto loadB = [&](int k0, int q[2], float sv[8]) {
        const int*   qrow = QW + (size_t)(k0 + kk2) * NPACKS + (col0 >> 3) + bp;
        const float* srow = S + (size_t)(k0 >> 7) * NDIM + col0 + bp * 8;
        q[0] = qrow[0];
        q[1] = qrow[NPACKS];
        float4 s0 = *reinterpret_cast<const float4*>(srow);
        float4 s1 = *reinterpret_cast<const float4*>(srow + 4);
        sv[0] = s0.x; sv[1] = s0.y; sv[2] = s0.z; sv[3] = s0.w;
        sv[4] = s1.x; sv[5] = s1.y; sv[6] = s1.z; sv[7] = s1.w;
    };

    // AWQ logical j -> nibble shift: order (0,4,1,5,2,6,3,7) * 4.
    // Sign-extend int4 via (q << (28-shift)) >> 28; pack 2 k per b32 store.
    const int awq_shift[8] = {0, 16, 4, 20, 8, 24, 12, 28};
    auto dequantStore = [&](int buf, const int q[2], const float sv[8]) {
        #pragma unroll
        for (int j = 0; j < 8; ++j) {
            int lsh = 28 - awq_shift[j];
            int v0 = ((int)((unsigned)q[0] << lsh)) >> 28;
            int v1 = ((int)((unsigned)q[1] << lsh)) >> 28;
            float f0 = (float)v0 * sv[j];
            float f1 = (float)v1 * sv[j];
            unsigned pk = (unsigned)f32_to_bf16(f0)
                        | ((unsigned)f32_to_bf16(f1) << 16);
            *reinterpret_cast<unsigned*>(&Bs[buf][bp * 8 + j][kk2]) = pk;
        }
    };

    // ---- Prologue: stage tile 0 into buffer 0 ----
    {
        stageA(0, 0);
        int q0[2]; float sv0[8];
        loadB(0, q0, sv0);
        dequantStore(0, q0, sv0);
        asm volatile("s_wait_asynccnt 0x0" ::: "memory");
        __syncthreads();
    }

    // 4x4 grid of 16x16 f32 accumulators per wave (64x64 output)
    v8f acc[4][4];
    #pragma unroll
    for (int mi = 0; mi < 4; ++mi)
        #pragma unroll
        for (int ni = 0; ni < 4; ++ni)
            acc[mi][ni] = v8f{0.f, 0.f, 0.f, 0.f, 0.f, 0.f, 0.f, 0.f};

    const int mrow = (lane & 15);
    const int kg   = (lane >> 4);

    int   qn[2];
    float svn[8];

    for (int kt = 0; kt < KTILES; ++kt) {
        const int  cur  = kt & 1;
        const int  nxt  = cur ^ 1;
        const bool more = (kt + 1 < KTILES);

        // Fire next tile's data movement before touching this tile's LDS.
        if (more) {
            stageA((kt + 1) * BK, nxt);     // async DMA -> LDS (other buffer)
            loadB((kt + 1) * BK, qn, svn);  // global -> regs
        }

        // ---- Gather WMMA fragments from LDS (current buffer) ----
        // A fragment 16x32 bf16: lane&15 = M row, lane>>4 = K half;
        // VGPR v holds K pair {kbase, kbase+1} -> two contiguous b128 per frag.
        union AFrag { v16bf v; unsigned u[8]; };
        AFrag af[4];
        #pragma unroll
        for (int mi = 0; mi < 4; ++mi) {
            int m = wm + mi * 16 + mrow;
            #pragma unroll
            for (int v = 0; v < 8; ++v) {
                int kk = ((v & 3) * 2) + kg * 8 + ((v >> 2) * 16);
                af[mi].u[v] = *reinterpret_cast<const unsigned*>(&As[cur][m][kk]);
            }
        }
        // B fragment 32x16 bf16: lane&15 = N col, lane>>4 = K half;
        // k {8h..8h+7, 16+8h..16+8h+7} per lane -> two b128 DS loads.
        union BFrag { v16bf v; uint4 q[2]; };
        BFrag bf[4];
        #pragma unroll
        for (int ni = 0; ni < 4; ++ni) {
            int n = wn + ni * 16 + mrow;
            bf[ni].q[0] = *reinterpret_cast<const uint4*>(&Bs[cur][n][kg * 8 + 0]);
            bf[ni].q[1] = *reinterpret_cast<const uint4*>(&Bs[cur][n][kg * 8 + 16]);
        }

        // ---- 16 WMMAs: D = A x B + C ----
        #pragma unroll
        for (int mi = 0; mi < 4; ++mi)
            #pragma unroll
            for (int ni = 0; ni < 4; ++ni)
                acc[mi][ni] = __builtin_amdgcn_wmma_f32_16x16x32_bf16(
                    false, af[mi].v, false, bf[ni].v,
                    (short)0, acc[mi][ni], false, false);

        // Dequant the register-staged B tile into the other buffer
        // (overlaps WMMA issue; waits only on its own loadcnt).
        if (more)
            dequantStore(nxt, qn, svn);

        asm volatile("s_wait_asynccnt 0x0" ::: "memory");
        __syncthreads();
    }

    // ---- Epilogue: bias add, bf16 convert, store ----
    // C/D layout: lane&15 = N col, VGPR i -> M row = i + 8*(lane>>4).
    #pragma unroll
    for (int ni = 0; ni < 4; ++ni) {
        int col = col0 + wn + ni * 16 + mrow;
        float bv = bf16_to_f32(bias[col]);
        #pragma unroll
        for (int mi = 0; mi < 4; ++mi) {
            #pragma unroll
            for (int i = 0; i < 8; ++i) {
                int row = row0 + wm + mi * 16 + i + 8 * kg;
                out[(size_t)row * NDIM + col] = f32_to_bf16(acc[mi][ni][i] + bv);
            }
        }
    }
}

extern "C" void kernel_launch(void* const* d_in, const int* in_sizes, int n_in,
                              void* d_out, int out_size, void* d_ws, size_t ws_size,
                              hipStream_t stream) {
    (void)n_in; (void)out_size; (void)d_ws; (void)ws_size;
    const unsigned short* X    = (const unsigned short*)d_in[0]; // bf16 bits
    const int*            QW   = (const int*)d_in[1];
    const float*          S    = (const float*)d_in[2];
    const unsigned short* bias = (const unsigned short*)d_in[3]; // bf16 bits
    unsigned short*       out  = (unsigned short*)d_out;         // bf16 bits

    int M = in_sizes[0] / KDIM;                 // B*T = 8192
    dim3 grid(NDIM / BN, M / BM);               // 86 x 32
    woq_awq_gemm_bf16<<<grid, 256, 0, stream>>>(X, QW, S, bias, out);
}
